// DirectMultiStepModel_62672162783861
// MI455X (gfx1250) — matrix operand order repaired and checked by reading, not compiled, not run on hardware
//
#include <hip/hip_runtime.h>

// ---------------------------------------------------------------------------
// Problem constants (match reference)
// ---------------------------------------------------------------------------
#define T_STEPS 24
#define NNODES  10000
#define NPAD    10112           // 79 * 128, zero-padded rows for 128-row tiles
#define D_IN    128
#define H1      256
#define H2      128
#define G1      (3*H1)          // 768
#define G2      (3*H2)          // 384
#define E_IN    320000
#define E_TOT   (E_IN + NNODES) // 330000 (self loops appended)
#define OUT_DIM 100

typedef __attribute__((ext_vector_type(16))) __bf16 v16bf;
typedef __attribute__((ext_vector_type(8)))  __bf16 v8bf;
typedef __attribute__((ext_vector_type(8)))  float  v8f;

// ---------------------------------------------------------------------------
// bf16 helpers (round-to-nearest-even)
// ---------------------------------------------------------------------------
__device__ __forceinline__ __bf16 f2bf(float f) {
  unsigned u = __float_as_uint(f);
  unsigned r = u + 0x7FFFu + ((u >> 16) & 1u);
  unsigned short s = (unsigned short)(r >> 16);
  return __builtin_bit_cast(__bf16, s);
}
__device__ __forceinline__ float bf2f(__bf16 b) {
  unsigned short s = __builtin_bit_cast(unsigned short, b);
  return __uint_as_float(((unsigned)s) << 16);
}

// ---------------------------------------------------------------------------
// zero workspace (grid-stride, 16B granules)
// ---------------------------------------------------------------------------
__global__ void zero_u4(uint4* p, size_t n16) {
  size_t i = (size_t)blockIdx.x * blockDim.x + threadIdx.x;
  size_t st = (size_t)gridDim.x * blockDim.x;
  uint4 z; z.x = z.y = z.z = z.w = 0u;
  for (; i < n16; i += st) p[i] = z;
}

// f32 -> bf16 bulk convert
__global__ __launch_bounds__(256) void conv_bf16(const float* __restrict__ s,
                                                 __bf16* __restrict__ d, int n) {
  int i = blockIdx.x * 256 + threadIdx.x;
  if (i < n) d[i] = f2bf(s[i]);
}

// ---------------------------------------------------------------------------
// WMMA bf16 GEMM:  C[M x NG] = A[M x K] * B[NG x K]^T   (f32 accumulate)
//   block = 128(M) x 64(N) tile, 128 threads = 4 waves (wave32)
//   wave w computes rows [w*32, w*32+32) x all 64 N columns:
//     2 A fragments x 4 B fragments -> 8 WMMA per k-step, B reused 2x.
//   Tiles staged memory->LDS with GLOBAL_LOAD_ASYNC_TO_LDS_B128 (ASYNCcnt),
//   avoiding the VGPR round-trip; explicit s_wait_asynccnt before barrier.
//   Fragment gathers follow CDNA5 16-bit A 16x32 layout (ISA 7.12.2).
// ---------------------------------------------------------------------------
__global__ __launch_bounds__(128)
void gemm_bf16_nt(const __bf16* __restrict__ A, const __bf16* __restrict__ B,
                  float* __restrict__ C, int K, int ldc) {
  __shared__ __align__(32) __bf16 lsA[128 * 32];
  __shared__ __align__(32) __bf16 lsB[64 * 32];

  const int tid  = threadIdx.x;
  const int lane = tid & 31;
  const int w    = tid >> 5;
  const int half = lane >> 4;   // 0: lanes 0-15, 1: lanes 16-31
  const int lr   = lane & 15;
  const int nbase = blockIdx.x * 64;
  const int mbase = blockIdx.y * 128;

  v8f acc0[4], acc1[4];
#pragma unroll
  for (int s = 0; s < 4; ++s)
#pragma unroll
    for (int i = 0; i < 8; ++i) { acc0[s][i] = 0.f; acc1[s][i] = 0.f; }

  for (int kk = 0; kk < K; kk += 32) {
    __syncthreads();   // previous iteration's LDS reads done before overwrite
    // async stage A tile 128x32 (512 x 16B chunks) and B tile 64x32 (256)
#pragma unroll
    for (int c = tid; c < 512; c += 128) {
      int rr = c >> 2;
      int ko = (c & 3) * 8;
      unsigned la = (unsigned)(size_t)&lsA[rr * 32 + ko];
      const __bf16* ga = A + (size_t)(mbase + rr) * K + kk + ko;
      asm volatile("global_load_async_to_lds_b128 %0, %1, off"
                   :: "v"(la), "v"(ga) : "memory");
    }
#pragma unroll
    for (int c = tid; c < 256; c += 128) {
      int rr = c >> 2;
      int ko = (c & 3) * 8;
      unsigned lb = (unsigned)(size_t)&lsB[rr * 32 + ko];
      const __bf16* gb = B + (size_t)(nbase + rr) * K + kk + ko;
      asm volatile("global_load_async_to_lds_b128 %0, %1, off"
                   :: "v"(lb), "v"(gb) : "memory");
    }
    asm volatile("s_wait_asynccnt 0x0" ::: "memory");
    __syncthreads();

    // A fragments: lane half 0 -> K 0..7 / 16..23 ; half 1 -> K 8..15 / 24..31
    const __bf16* arow0 = &lsA[(w * 32      + lr) * 32 + half * 8];
    const __bf16* arow1 = &lsA[(w * 32 + 16 + lr) * 32 + half * 8];
    v8bf a0lo = *(const v8bf*)(arow0);
    v8bf a0hi = *(const v8bf*)(arow0 + 16);
    v8bf a1lo = *(const v8bf*)(arow1);
    v8bf a1hi = *(const v8bf*)(arow1 + 16);
    v16bf af0, af1;
#pragma unroll
    for (int i = 0; i < 8; ++i) {
      af0[i] = a0lo[i]; af0[8 + i] = a0hi[i];
      af1[i] = a1lo[i]; af1[8 + i] = a1hi[i];
    }

#pragma unroll
    for (int s = 0; s < 4; ++s) {
      // B fragment: column n = s*16+lr; half 0 -> K 0..7 / 16..23 ; half 1 -> K 8..15 / 24..31
      const __bf16* brow = &lsB[(s * 16 + lr) * 32 + half * 8];
      v8bf blo = *(const v8bf*)(brow);
      v8bf bhi = *(const v8bf*)(brow + 16);
      v16bf bf_;
#pragma unroll
      for (int i = 0; i < 8; ++i) { bf_[i] = blo[i]; bf_[8 + i] = bhi[i]; }
      acc0[s] = __builtin_amdgcn_wmma_f32_16x16x32_bf16(
          false, af0, false, bf_, (short)0, acc0[s], false, false);
      acc1[s] = __builtin_amdgcn_wmma_f32_16x16x32_bf16(
          false, af1, false, bf_, (short)0, acc1[s], false, false);
    }
  }

  // C/D layout (ISA 7.12.2): VGPR i -> M = half*8 + i, N = lr
  const int crow0 = mbase + w * 32 + half * 8;
  const int ccol0 = nbase + lr;
#pragma unroll
  for (int s = 0; s < 4; ++s) {
    float* cp0 = C + (size_t)crow0 * ldc + ccol0 + s * 16;
    float* cp1 = cp0 + (size_t)16 * ldc;
#pragma unroll
    for (int i = 0; i < 8; ++i) {
      cp0[(size_t)i * ldc] = acc0[s][i];
      cp1[(size_t)i * ldc] = acc1[s][i];
    }
  }
}

// ---------------------------------------------------------------------------
// GRU gate fusion: h_new = (1-z)*n + z*h   (torch nn.GRU semantics)
// ---------------------------------------------------------------------------
__global__ __launch_bounds__(256)
void gru_gates(const float* __restrict__ gi, const float* __restrict__ gh,
               const float* __restrict__ bih, const float* __restrict__ bhh,
               const __bf16* __restrict__ hprev, __bf16* __restrict__ hnew,
               int H, int Nreal) {
  int idx = blockIdx.x * 256 + threadIdx.x;
  if (idx >= Nreal * H) return;
  int m = idx / H, j = idx - m * H;
  int G = 3 * H;
  size_t base = (size_t)m * G + j;
  float ir = gi[base]         + bih[j];
  float iz = gi[base + H]     + bih[H + j];
  float in_ = gi[base + 2*H]  + bih[2*H + j];
  float hr = gh[base]         + bhh[j];
  float hz = gh[base + H]     + bhh[H + j];
  float hn = gh[base + 2*H]   + bhh[2*H + j];
  float r = 1.f / (1.f + expf(-(ir + hr)));
  float z = 1.f / (1.f + expf(-(iz + hz)));
  float n = tanhf(in_ + r * hn);
  float hp = bf2f(hprev[(size_t)m * H + j]);
  hnew[(size_t)m * H + j] = f2bf((1.f - z) * n + z * hp);
}

// ---------------------------------------------------------------------------
// Degree / symmetric norm
// ---------------------------------------------------------------------------
__global__ void deg_init(float* deg, int n) {
  int i = blockIdx.x * 256 + threadIdx.x;
  if (i < n) deg[i] = 1.0f;   // self-loop contributes 1
}
__global__ void deg_acc(const int* __restrict__ ei, float* deg, int E) {
  int e = blockIdx.x * 256 + threadIdx.x;
  if (e < E) atomicAdd(&deg[ei[E + e]], 1.0f);   // segment over col
}
__global__ void norm_k(const int* __restrict__ ei, const float* __restrict__ deg,
                       float* __restrict__ nv, int E, int Etot) {
  int e = blockIdx.x * 256 + threadIdx.x;
  if (e >= Etot) return;
  if (e < E) nv[e] = rsqrtf(deg[ei[e]]) * rsqrtf(deg[ei[E + e]]);
  else       nv[e] = 1.0f / deg[e - E];          // self loop: dis^2
}

// ---------------------------------------------------------------------------
// Edge gather / scatter-add:  agg[col] += norm_e * h[row]   (8 feats/thread)
// ---------------------------------------------------------------------------
__global__ __launch_bounds__(256)
void edge_agg(const int* __restrict__ ei, const float* __restrict__ normv,
              const __bf16* __restrict__ hsrc, float* __restrict__ agg,
              int E, int Etot, int H) {
  int chunks = H >> 3;
  long long idx = (long long)blockIdx.x * 256 + threadIdx.x;
  if (idx >= (long long)Etot * chunks) return;
  int e = (int)(idx / chunks);
  int c = ((int)(idx - (long long)e * chunks)) << 3;
  int row, col;
  if (e < E) { row = ei[e]; col = ei[E + e]; } else { row = e - E; col = row; }
  float nm = normv[e];
  v8bf v = *(const v8bf*)(hsrc + (size_t)row * H + c);
  float* dst = agg + (size_t)col * H + c;
#pragma unroll
  for (int i = 0; i < 8; ++i) atomicAdd(dst + i, nm * bf2f(v[i]));
}

// relu(agg + bias) -> bf16 (layer-2 input slice)
__global__ __launch_bounds__(256)
void relu_bias_bf16(const float* __restrict__ agg, const float* __restrict__ bias,
                    __bf16* __restrict__ dst, int H, int Nreal) {
  int i = blockIdx.x * 256 + threadIdx.x;
  if (i >= Nreal * H) return;
  float v = agg[i] + bias[i & (H - 1)];
  dst[i] = f2bf(v > 0.f ? v : 0.f);
}

// ---------------------------------------------------------------------------
// Output head: logits[o] = (agg2 + bias2) . lin_W[o]   (512 MB HBM stream)
// 100 outputs x 8 chunk-blocks, block reduce + atomic combine
// ---------------------------------------------------------------------------
__global__ __launch_bounds__(256)
void linear_partial(const float* __restrict__ agg2, const float* __restrict__ bias2,
                    const float* __restrict__ Wl, float* __restrict__ logits) {
  __shared__ float sh[256];
  const int o = blockIdx.x >> 3;
  const int chunk = blockIdx.x & 7;
  const int t = threadIdx.x;
  const size_t L = (size_t)NNODES * H2;       // 1,280,000
  const size_t seg = L / 8;                   // 160,000
  const float* wrow = Wl + (size_t)o * L;
  float p = 0.f;
  size_t end = (size_t)(chunk + 1) * seg;
  for (size_t i = (size_t)chunk * seg + t; i < end; i += 256)
    p += (agg2[i] + bias2[i & (H2 - 1)]) * wrow[i];
  sh[t] = p; __syncthreads();
  for (int s = 128; s > 0; s >>= 1) { if (t < s) sh[t] += sh[t + s]; __syncthreads(); }
  if (t == 0) atomicAdd(&logits[o], sh[0]);
}

__global__ void softmax100(const float* __restrict__ logits,
                           const float* __restrict__ lin_b, float* __restrict__ out) {
  __shared__ float sh[128];
  int t = threadIdx.x;
  float v = (t < OUT_DIM) ? (logits[t] + lin_b[t]) : -3.0e38f;
  sh[t] = v; __syncthreads();
  for (int s = 64; s > 0; s >>= 1) { if (t < s) sh[t] = fmaxf(sh[t], sh[t + s]); __syncthreads(); }
  float mx = sh[0]; __syncthreads();
  float e = (t < OUT_DIM) ? expf(v - mx) : 0.f;
  sh[t] = e; __syncthreads();
  for (int s = 64; s > 0; s >>= 1) { if (t < s) sh[t] += sh[t + s]; __syncthreads(); }
  float sum = sh[0];
  if (t < OUT_DIM) out[t] = e / sum;
}

// ---------------------------------------------------------------------------
// kernel_launch
// ---------------------------------------------------------------------------
extern "C" void kernel_launch(void* const* d_in, const int* in_sizes, int n_in,
                              void* d_out, int out_size, void* d_ws, size_t ws_size,
                              hipStream_t stream) {
  (void)in_sizes; (void)n_in; (void)out_size; (void)ws_size;

  const float* x     = (const float*)d_in[0];
  const int*   ei    = (const int*)d_in[1];
  const float* W_ih1 = (const float*)d_in[2];
  const float* W_hh1 = (const float*)d_in[3];
  const float* b_ih1 = (const float*)d_in[4];
  const float* b_hh1 = (const float*)d_in[5];
  const float* bias1 = (const float*)d_in[6];
  const float* W_ih2 = (const float*)d_in[7];
  const float* W_hh2 = (const float*)d_in[8];
  const float* b_ih2 = (const float*)d_in[9];
  const float* b_hh2 = (const float*)d_in[10];
  const float* bias2 = (const float*)d_in[11];
  const float* lin_W = (const float*)d_in[12];
  const float* lin_b = (const float*)d_in[13];

  // ---- workspace carve-up (~222 MB) ----
  char* w = (char*)d_ws;
  size_t off = 0;
  auto take = [&](size_t bytes) -> void* {
    void* p = w + off;
    off = (off + bytes + 255) & ~(size_t)255;
    return p;
  };
  __bf16* wih1b = (__bf16*)take((size_t)G1 * D_IN * 2);
  __bf16* whh1b = (__bf16*)take((size_t)G1 * H1   * 2);
  __bf16* wih2b = (__bf16*)take((size_t)G2 * H1   * 2);
  __bf16* whh2b = (__bf16*)take((size_t)G2 * H2   * 2);
  __bf16* xb    = (__bf16*)take((size_t)NPAD * D_IN * 2);            // per-t x slice
  __bf16* h1b   = (__bf16*)take((size_t)(T_STEPS + 1) * NPAD * H1 * 2); // all h1 states
  __bf16* h2b   = (__bf16*)take((size_t)2 * NPAD * H2 * 2);          // ping-pong
  __bf16* in2b  = (__bf16*)take((size_t)NPAD * H1 * 2);              // relu slice
  float*  agg1  = (float*)take((size_t)NPAD * H1 * 4);               // per-t agg slice
  float*  agg2  = (float*)take((size_t)NPAD * H2 * 4);
  float*  giS   = (float*)take((size_t)NPAD * G1 * 4);               // gate GEMM out
  float*  ghS   = (float*)take((size_t)NPAD * G1 * 4);
  float*  deg   = (float*)take((size_t)NNODES * 4);
  float*  normv = (float*)take((size_t)E_TOT * 4);
  float*  logits= (float*)take(256 * 4);
  const size_t total = off;

  auto nb = [](long long n) { return (unsigned)((n + 255) / 256); };
  const size_t S1 = (size_t)NPAD * H1;   // h1 slice stride (elements)
  const size_t S2 = (size_t)NPAD * H2;

  // 1) zero all used workspace (pads, h0 slices, accumulators) each call
  zero_u4<<<8192, 256, 0, stream>>>((uint4*)d_ws, total / 16);

  // 2) weights -> bf16
  conv_bf16<<<nb(G1 * D_IN), 256, 0, stream>>>(W_ih1, wih1b, G1 * D_IN);
  conv_bf16<<<nb(G1 * H1),   256, 0, stream>>>(W_hh1, whh1b, G1 * H1);
  conv_bf16<<<nb(G2 * H1),   256, 0, stream>>>(W_ih2, wih2b, G2 * H1);
  conv_bf16<<<nb(G2 * H2),   256, 0, stream>>>(W_hh2, whh2b, G2 * H2);

  // 3) degrees + symmetric norm
  deg_init<<<nb(NNODES), 256, 0, stream>>>(deg, NNODES);
  deg_acc<<<nb(E_IN), 256, 0, stream>>>(ei, deg, E_IN);
  norm_k<<<nb(E_TOT), 256, 0, stream>>>(ei, deg, normv, E_IN, E_TOT);

  // 4) layer-1 GRU recurrence (24 sequential steps, all WMMA GEMMs)
  for (int t = 0; t < T_STEPS; ++t) {
    conv_bf16<<<nb((long long)NNODES * D_IN), 256, 0, stream>>>(
        x + (size_t)t * NNODES * D_IN, xb, NNODES * D_IN);
    gemm_bf16_nt<<<dim3(G1 / 64, NPAD / 128), 128, 0, stream>>>(
        xb, wih1b, giS, D_IN, G1);                               // gi = x_t @ W_ih1^T
    gemm_bf16_nt<<<dim3(G1 / 64, NPAD / 128), 128, 0, stream>>>(
        h1b + (size_t)t * S1, whh1b, ghS, H1, G1);               // gh = h_{t-1} @ W_hh1^T
    gru_gates<<<nb((long long)NNODES * H1), 256, 0, stream>>>(
        giS, ghS, b_ih1, b_hh1, h1b + (size_t)t * S1, h1b + (size_t)(t + 1) * S1,
        H1, NNODES);
  }

  // 5) per-timestep: layer-1 graph aggregation -> relu -> layer-2 GRU step
  for (int t = 0; t < T_STEPS; ++t) {
    zero_u4<<<2048, 256, 0, stream>>>((uint4*)agg1, ((size_t)NPAD * H1 * 4) / 16);
    edge_agg<<<nb((long long)E_TOT * (H1 / 8)), 256, 0, stream>>>(
        ei, normv, h1b + (size_t)(t + 1) * S1, agg1, E_IN, E_TOT, H1);
    relu_bias_bf16<<<nb((long long)NNODES * H1), 256, 0, stream>>>(
        agg1, bias1, in2b, H1, NNODES);

    const __bf16* h2p = h2b + (size_t)(t & 1) * S2;
    __bf16*       h2n = h2b + (size_t)((t + 1) & 1) * S2;
    gemm_bf16_nt<<<dim3(G2 / 64, NPAD / 128), 128, 0, stream>>>(
        in2b, wih2b, giS, H1, G2);                               // gi2
    gemm_bf16_nt<<<dim3(G2 / 64, NPAD / 128), 128, 0, stream>>>(
        h2p, whh2b, ghS, H2, G2);                                // gh2
    gru_gates<<<nb((long long)NNODES * H2), 256, 0, stream>>>(
        giS, ghS, b_ih2, b_hh2, h2p, h2n, H2, NNODES);
  }

  // 6) layer-2 aggregation, last timestep only (agg2 zeroed by step 1)
  const __bf16* h2fin = h2b + (size_t)(T_STEPS & 1) * S2;
  edge_agg<<<nb((long long)E_TOT * (H2 / 8)), 256, 0, stream>>>(
      ei, normv, h2fin, agg2, E_IN, E_TOT, H2);

  // 7) output head: 512 MB lin_W stream (~22 us roofline), then softmax
  linear_partial<<<OUT_DIM * 8, 256, 0, stream>>>(agg2, bias2, lin_W, logits);
  softmax100<<<1, 128, 0, stream>>>(logits, lin_b, (float*)d_out);
}
